// DolphinBrain_15195594293724
// MI455X (gfx1250) — compile-verified, matrix operand order checked by reading, not run on hardware
//
#include <hip/hip_runtime.h>
#include <hip/hip_bf16.h>
#include <math.h>

// ---------------------------------------------------------------------------
// Model dimensions (from reference)
// ---------------------------------------------------------------------------
constexpr int Bb = 128, Tt = 1024, INs = 4, Hh = 256;
constexpr int AUD=25, SEN=15, BRA=15, LCX=46, RCX=46, CAL=20, PAR=25, HIP=15, CER=25, MOT=20;
// state layout == REGION_ORDER concat (tight), padded to 256 for K-chunks of 32
constexpr int O_AUD=0, O_SEN=25, O_BRA=40, O_LCX=55, O_RCX=101, O_CAL=147,
              O_PAR=167, O_HIP=192, O_CER=207, O_MOT=232;
constexpr int STW = 256;

// Output layout: (predictions, hidden, anomaly, surprise, props) concat, flat f32
constexpr size_t OUT_PRED = 0;
constexpr size_t OUT_HID  = (size_t)Bb*(Tt-1)*INs;
constexpr size_t OUT_ANOM = OUT_HID + (size_t)Bb*Tt*Hh;
constexpr size_t OUT_SUR  = OUT_ANOM + (size_t)Bb*Tt;
constexpr size_t OUT_PROP = OUT_SUR  + (size_t)Bb*(Tt-1);

// ---------------------------------------------------------------------------
// Flattened input index map (insertion-order flattening of setup_inputs())
// ---------------------------------------------------------------------------
#define NIN 103
enum {
  I_X=0, I_IPROJ_W=1, I_IPROJ_B=2,
  R_AUD=3, R_SEN=10, R_BRA=17, R_LCX=24, R_RCX=31, R_CAL=38, R_PAR=45, R_HIP=52,
  R_CER=59, R_MOT=66,
  I_P2C=73, I_H2A=77, I_SW_W=81, I_SW_B=82,
  I_THAL_W=83, I_THAL_B=84, I_THAL_LNG=85, I_THAL_LNB=86,
  I_TPROJ_W=87, I_TPROJ_B=88,
  I_NEXT_LNG=89, I_NEXT_LNB=90, I_NEXT_W=91, I_NEXT_B=92,
  I_ANOM_LNG=93, I_ANOM_LNB=94, I_ANOM_W=95, I_ANOM_B=96,
  I_PROP_LNG=97, I_PROP_LNB=98, I_P1_W=99, I_P1_B=100, I_P2_W=101, I_P2_B=102
};
struct PtrTab { const float* p[NIN]; };

// ---------------------------------------------------------------------------
// Packed f16 weights in d_ws, stored in *WMMA fragment order*:
//   element index = ((nt*Kc + kc)*32 + lane)*16 + j
// so a B-fragment is one aligned 32-byte v16h load per lane.
// Zero-padded K rows make junk A columns past the true K harmless.
// ---------------------------------------------------------------------------
constexpr int NW = 28;
constexpr int WIdx[NW] = {1,3,5,10,12,17,19,24,26,31,33,38,40,45,47,52,54,59,61,66,68,73,75,77,79,81,83,87};
constexpr int WK[NW]   = {4,25,25,25,15,40,15,101,46,101,46,92,20,137,25,107,15,61,25,117,20,25,25,15,15,15,252,252};
constexpr int WN[NW]   = {25,25,25,15,15,15,15,46,46,46,46,20,20,25,25,15,15,25,25,20,20,55,55,25,25,2,256,256};
constexpr int WKp[NW]  = {32,32,32,32,32,64,32,128,64,128,64,96,32,160,32,128,32,64,32,128,32,32,32,32,32,32,256,256};
constexpr int WNp[NW]  = {32,32,32,16,16,16,16,48,48,48,48,32,32,32,32,16,16,32,32,32,32,64,64,32,32,16,256,256};
constexpr int WOFFS[NW]= {0,1024,2048,3072,3584,4096,5120,5632,11776,14848,20992,24064,27136,28160,33280,
                          34304,36352,36864,38912,39936,44032,45056,47104,49152,50176,51200,51712,117248};
// total = 182784 f16 elements (~357 KB of d_ws)

typedef __attribute__((ext_vector_type(16))) _Float16 v16h;
typedef __attribute__((ext_vector_type(8)))  float    v8f;

// K-index striping for 16-bit A/B fragments (ISA 7.12.2: lanes 0-15 hold
// K{0-7,16-23}, lanes 16-31 hold K{8-15,24-31} across the 8 VGPR pairs).
__device__ __forceinline__ int kmap(int lane, int j) {
  return ((j >> 3) << 4) | (((lane >> 4) & 1) << 3) | (j & 7);
}

// One 16x16x32 MMA: A gathered from LDS (stride ldA), B = fragment-packed
// global weights (single v16h load).
__device__ __forceinline__ v8f frag_mma(const _Float16* __restrict__ A, int ldA, int kc,
                                        const _Float16* __restrict__ Bf, int tile,
                                        int lane, v8f acc)
{
  const int m = lane & 15;
  v16h va;
#pragma unroll
  for (int j = 0; j < 16; ++j) va[j] = A[m*ldA + kc*32 + kmap(lane, j)];
  v16h vb = ((const v16h*)Bf)[tile*32 + lane];
  return __builtin_amdgcn_wmma_f32_16x16x32_f16(false, va, false, vb,
                                                (short)0, acc, false, false);
}

// D(16 x ntiles*16) = A(16 x Kc*32) x Bf, written to f32 tile P (stride ldP)
__device__ __forceinline__ void gemm1_wmma(const _Float16* __restrict__ A, int ldA, int Kc,
                                           const _Float16* __restrict__ Bf, int ntiles,
                                           float* __restrict__ P, int ldP, int lane)
{
  const int m = lane & 15, kh = (lane >> 4) & 1;
  for (int nt = 0; nt < ntiles; ++nt) {
    v8f acc;
#pragma unroll
    for (int r = 0; r < 8; ++r) acc[r] = 0.f;
    for (int kc = 0; kc < Kc; ++kc) acc = frag_mma(A, ldA, kc, Bf, nt*Kc + kc, lane, acc);
#pragma unroll
    for (int r = 0; r < 8; ++r) P[(r + 8*kh)*ldP + nt*16 + m] = acc[r];
  }
}

// Fused in+rec GEMM (register-resident accumulator) + liquid cell + LayerNorm.
__device__ __forceinline__ void region_step_dev(
    const _Float16* __restrict__ A, int ldA, int KcIn, int wIn,
    _Float16* SB, int selfOff, int recKc, int wRec,
    int ntiles, int Nreal, const PtrTab& pt, int pbase, float dt,
    float* __restrict__ P, int lane, const _Float16* __restrict__ ws)
{
  const int m = lane & 15, kh = (lane >> 4) & 1;
  const _Float16* B1 = ws + WOFFS[wIn];
  const _Float16* B2 = ws + WOFFS[wRec];
  const _Float16* A2 = SB + selfOff;
  for (int nt = 0; nt < ntiles; ++nt) {
    v8f acc;
#pragma unroll
    for (int r = 0; r < 8; ++r) acc[r] = 0.f;
    for (int kc = 0; kc < KcIn;  ++kc) acc = frag_mma(A,  ldA, kc, B1, nt*KcIn  + kc, lane, acc);
    for (int kc = 0; kc < recKc; ++kc) acc = frag_mma(A2, STW, kc, B2, nt*recKc + kc, lane, acc);
#pragma unroll
    for (int r = 0; r < 8; ++r) P[(r + 8*kh)*64 + nt*16 + m] = acc[r];
  }
  if (lane < 16) {
    const int row = lane;
    const float* inb = pt.p[pbase+1];
    const float* reb = pt.p[pbase+3];
    const float* ls  = pt.p[pbase+4];
    const float* lg  = pt.p[pbase+5];
    const float* lb  = pt.p[pbase+6];
    _Float16* hp = SB + row*STW + selfOff;
    float* pr = P + row*64;
    float mean = 0.f;
    for (int f = 0; f < Nreal; ++f) {
      float pre = pr[f] + inb[f] + reb[f];
      float s   = expf(ls[f]) * dt;
      float hn  = ((float)hp[f] + s * tanhf(pre)) / (1.f + s);
      pr[f] = hn; mean += hn;
    }
    mean /= (float)Nreal;
    float var = 0.f;
    for (int f = 0; f < Nreal; ++f) { float d = pr[f]-mean; var += d*d; }
    var /= (float)Nreal;
    float inv = rsqrtf(var + 1e-5f);
    for (int f = 0; f < Nreal; ++f)
      hp[f] = (_Float16)((pr[f]-mean)*inv*lg[f] + lb[f]);
  }
}

// ---------------------------------------------------------------------------
// Phase 0: pack f32 weights -> fragment-swizzled padded f16 in d_ws
// ---------------------------------------------------------------------------
__global__ void pack_weights_kernel(PtrTab pt, _Float16* __restrict__ ws)
{
  int widx = blockIdx.x;
  if (widx >= NW) return;
  const float* src = pt.p[WIdx[widx]];
  const int K = WK[widx], N = WN[widx];
  const int Kc = WKp[widx] >> 5;
  const int total = WKp[widx]*WNp[widx];
  _Float16* dst = ws + WOFFS[widx];
  for (int e = threadIdx.x; e < total; e += blockDim.x) {
    int j    = e & 15;
    int lane = (e >> 4) & 31;
    int tile = e >> 9;
    int kc   = tile % Kc;
    int nt   = tile / Kc;
    int k = kc*32 + kmap(lane, j);
    int n = nt*16 + (lane & 15);
    float v = (k < K && n < N) ? src[k*N + n] : 0.f;
    dst[e] = (_Float16)v;
  }
}

// ---------------------------------------------------------------------------
// Phase 1: persistent recurrent kernel. 1 workgroup, 8 wave32s; wave w owns
// batch rows [16w, 16w+16). All state lives in LDS across T=1024 steps.
// ---------------------------------------------------------------------------
__global__ __launch_bounds__(256, 1)
void brain_rnn_kernel(PtrTab pt, const _Float16* __restrict__ ws, float* __restrict__ out)
{
  extern __shared__ char smem[];
  _Float16* state  = (_Float16*)smem;                               // 128*256 f16
  float*    preAll = (float*)(smem + (size_t)Bb*STW*2);             // 8*16*64 f32
  _Float16* stgAll = (_Float16*)((char*)preAll + 8*16*64*4);        // 8*16*128 f16
  _Float16* tprAll = (_Float16*)((char*)stgAll + 8*16*128*2);       // 8*16*256 f16
  float*    red    = (float*)((char*)tprAll + 8*16*256*2);          // 16 f32
  float*    xstgA  = red + 16;                                      // 8*16*4 f32

  const int tid  = threadIdx.x;
  const int w    = tid >> 5;
  const int lane = tid & 31;
  const int m0   = w * 16;
  const int m    = lane & 15;
  const int kh   = (lane >> 4) & 1;
  float*    P    = preAll + w*16*64;
  _Float16* stg  = stgAll + w*16*128;
  _Float16* tpre = tprAll + w*16*256;
  _Float16* tscr = tpre;               // 16x64 scratch, free until thalamus phase
  float*    xstg = xstgA + w*64;
  _Float16* SB   = state + m0*STW;

  for (int i = tid; i < Bb*STW; i += 256) state[i] = (_Float16)0.f;
  __syncthreads();

  const float* x      = pt.p[I_X];
  const float* iprojb = pt.p[I_IPROJ_B];
  const float* h2a_pb = pt.p[I_H2A+1];
  const float* h2a_gb = pt.p[I_H2A+3];
  const float* p2c_pb = pt.p[I_P2C+1];
  const float* p2c_gb = pt.p[I_P2C+3];
  const float* swb    = pt.p[I_SW_B];
  const float* thalb  = pt.p[I_THAL_B];
  const float* tlng   = pt.p[I_THAL_LNG];
  const float* tlnb   = pt.p[I_THAL_LNB];
  const float* tpb    = pt.p[I_TPROJ_B];

  for (int t = 0; t < Tt; ++t) {
    const float mask = (t > 0) ? 1.f : 0.f;
    if (lane == 0) {      // keep the big thalamus weights hot (global_prefetch_b8)
      __builtin_prefetch(ws + WOFFS[26], 0, 2);
      __builtin_prefetch(ws + WOFFS[27], 0, 2);
    }

    // ---- aud_in = input_proj(x_t) + 0.1*mask*axon(h2a, hippo_old) ----
    gemm1_wmma(SB + O_HIP, STW, 1, ws + WOFFS[23], 2, P, 64, lane);
    if (lane < 16) for (int f = 0; f < AUD; ++f)
      tscr[lane*64+f] = (_Float16)tanhf(P[lane*64+f] + h2a_pb[f]);
    gemm1_wmma(SB + O_HIP, STW, 1, ws + WOFFS[24], 2, P, 64, lane);
    if (lane < 16) for (int f = 0; f < AUD; ++f) {
      float g = 1.f/(1.f+expf(-(P[lane*64+f] + h2a_gb[f])));
      tscr[lane*64+f] = (_Float16)(0.1f * mask * g * (float)tscr[lane*64+f]);
    }
    // x tile -> LDS via CDNA5 async DMA (GLOBAL_LOAD_ASYNC_TO_LDS_B128):
    // per-lane VDST VGPR carries the wave-relative LDS byte offset (= low 32
    // bits of the generic LDS pointer), VADDR the 64-bit global address.
    if (lane < 16) {
      const float* xr = x + ((size_t)(m0+lane))*Tt*INs + (size_t)t*INs;
      unsigned     lds_off = (unsigned)(unsigned long long)(uintptr_t)(xstg + lane*4);
      unsigned long long ga = (unsigned long long)(uintptr_t)xr;
      asm volatile("global_load_async_to_lds_b128 %0, %1, off"
                   :: "v"(lds_off), "v"(ga) : "memory");
    }
    asm volatile("s_wait_asynccnt 0x0" ::: "memory");
    if (lane < 16) {
      _Float16* sg = stg + lane*128;
      for (int c = 0; c < INs; ++c) sg[c] = (_Float16)xstg[lane*4 + c];
      for (int c = INs; c < 32; ++c) sg[c] = (_Float16)0.f;
    }
    gemm1_wmma(stg, 128, 1, ws + WOFFS[0], 2, P, 64, lane);
    if (lane < 16) {
      _Float16* sg = stg + lane*128;
      for (int f = 0; f < AUD; ++f)
        sg[f] = (_Float16)(P[lane*64+f] + iprojb[f] + (float)tscr[lane*64+f]);
      for (int f = AUD; f < 32; ++f) sg[f] = (_Float16)0.f;
    }
    // auditory / sensory / brainstem (latter two read state slices directly)
    region_step_dev(stg,       128, 1, 1,  SB, O_AUD, 1, 2,  2, AUD, pt, R_AUD, 2.5f, P, lane, ws);
    region_step_dev(SB+O_AUD,  STW, 1, 3,  SB, O_SEN, 1, 4,  1, SEN, pt, R_SEN, 1.5f, P, lane, ws);
    region_step_dev(SB+O_AUD,  STW, 2, 5,  SB, O_BRA, 1, 6,  1, BRA, pt, R_BRA, 2.0f, P, lane, ws);

    // switch gate -> batch-mean dt scales (only cross-wave coupling per step)
    gemm1_wmma(SB + O_BRA, STW, 1, ws + WOFFS[25], 1, P, 64, lane);
    if (lane == 0) { red[w*2+0] = 0.f; red[w*2+1] = 0.f; }
    if (lane < 16) {
      float z0 = P[lane*64+0] + swb[0];
      float z1 = P[lane*64+1] + swb[1];
      float mx = fmaxf(z0, z1);
      float e0 = expf(z0-mx), e1 = expf(z1-mx);
      float p0 = e0 / (e0 + e1);
      atomicAdd(&red[w*2+0], 0.5f + p0);
      atomicAdd(&red[w*2+1], 0.5f + (1.f - p0));
    }
    __syncthreads();
    float ldt = 0.f, rdt = 0.f;
    for (int i = 0; i < 8; ++i) { ldt += red[i*2+0]; rdt += red[i*2+1]; }
    ldt /= (float)Bb; rdt /= (float)Bb;
    __syncthreads();

    // ctx_in common (gather [sen|aud|bra]) + 0.05*mask*axon(p2c, para_old)
    gemm1_wmma(SB + O_PAR, STW, 1, ws + WOFFS[21], 4, P, 64, lane);
    if (lane < 16) for (int f = 0; f < 55; ++f)
      tscr[lane*64+f] = (_Float16)tanhf(P[lane*64+f] + p2c_pb[f]);
    gemm1_wmma(SB + O_PAR, STW, 1, ws + WOFFS[22], 4, P, 64, lane);
    if (lane < 16) {
      const _Float16* srow = SB + lane*STW;
      _Float16* sg = stg + lane*128;
      for (int f = 0; f < 55; ++f) {
        float sv = (f < 15) ? (float)srow[O_SEN+f]
                 : (f < 40) ? (float)srow[O_AUD+(f-15)]
                            : (float)srow[O_BRA+(f-40)];
        float g = 1.f/(1.f+expf(-(P[lane*64+f] + p2c_gb[f])));
        sg[f] = (_Float16)(sv + 0.05f*mask*g*(float)tscr[lane*64+f]);
      }
      for (int f = 101; f < 128; ++f) sg[f] = (_Float16)0.f;
    }
    // left / right ctx (shared staged common + own old h appended)
    if (lane < 16) { const _Float16* srow = SB + lane*STW; _Float16* sg = stg + lane*128;
      for (int f = 0; f < LCX; ++f) sg[55+f] = srow[O_LCX+f]; }
    region_step_dev(stg, 128, 4, 7,  SB, O_LCX, 2, 8,  3, LCX, pt, R_LCX, ldt*0.3f, P, lane, ws);
    if (lane < 16) { const _Float16* srow = SB + lane*STW; _Float16* sg = stg + lane*128;
      for (int f = 0; f < RCX; ++f) sg[55+f] = srow[O_RCX+f]; }
    region_step_dev(stg, 128, 4, 9,  SB, O_RCX, 2, 10, 3, RCX, pt, R_RCX, rdt*0.3f, P, lane, ws);
    // callosum: [lcx|rcx] contiguous; paralimbic: [lcx|rcx|cal|self] contiguous
    region_step_dev(SB+O_LCX, STW, 3, 11, SB, O_CAL, 1, 12, 2, CAL, pt, R_CAL, 1.0f, P, lane, ws);
    region_step_dev(SB+O_LCX, STW, 5, 13, SB, O_PAR, 1, 14, 2, PAR, pt, R_PAR, 0.5f, P, lane, ws);
    // hippo: staged [lcx|rcx | hip_old]
    if (lane < 16) { const _Float16* srow = SB + lane*STW; _Float16* sg = stg + lane*128;
      for (int f = 0; f < 92; ++f)  sg[f]     = srow[O_LCX+f];
      for (int f = 0; f < HIP; ++f) sg[92+f]  = srow[O_HIP+f];
      for (int f = 107; f < 128; ++f) sg[f]   = (_Float16)0.f; }
    region_step_dev(stg, 128, 4, 15, SB, O_HIP, 1, 16, 1, HIP, pt, R_HIP, 0.7f, P, lane, ws);
    // cerebellum: staged [sen|lcx]
    if (lane < 16) { const _Float16* srow = SB + lane*STW; _Float16* sg = stg + lane*128;
      for (int f = 0; f < SEN; ++f) sg[f]      = srow[O_SEN+f];
      for (int f = 0; f < LCX; ++f) sg[SEN+f]  = srow[O_LCX+f];
      for (int f = 61; f < 64; ++f) sg[f]      = (_Float16)0.f; }
    region_step_dev(stg, 128, 2, 17, SB, O_CER, 1, 18, 2, CER, pt, R_CER, 1.8f, P, lane, ws);
    // motor: staged [lcx|rcx|cer]
    if (lane < 16) { const _Float16* srow = SB + lane*STW; _Float16* sg = stg + lane*128;
      for (int f = 0; f < 92; ++f)  sg[f]    = srow[O_LCX+f];
      for (int f = 0; f < CER; ++f) sg[92+f] = srow[O_CER+f];
      for (int f = 117; f < 128; ++f) sg[f]  = (_Float16)0.f; }
    region_step_dev(stg, 128, 4, 19, SB, O_MOT, 1, 20, 2, MOT, pt, R_MOT, 1.5f, P, lane, ws);

    // ---- thalamus gate: sigmoid(LN(allr @ Wthal + b)) ----
    {
      const _Float16* Bw = ws + WOFFS[26];
      for (int nt = 0; nt < 16; ++nt) {
        v8f acc;
#pragma unroll
        for (int r = 0; r < 8; ++r) acc[r] = 0.f;
        for (int kc = 0; kc < 8; ++kc) acc = frag_mma(SB, STW, kc, Bw, nt*8 + kc, lane, acc);
#pragma unroll
        for (int r = 0; r < 8; ++r)
          tpre[(r + 8*kh)*256 + nt*16 + m] = (_Float16)acc[r];
      }
    }
    if (lane < 16) {
      _Float16* tp = tpre + lane*256;
      float mean = 0.f;
      for (int f = 0; f < Hh; ++f) { float v = (float)tp[f] + thalb[f]; tp[f] = (_Float16)v; mean += v; }
      mean /= (float)Hh;
      float var = 0.f;
      for (int f = 0; f < Hh; ++f) { float d = (float)tp[f]-mean; var += d*d; }
      var /= (float)Hh;
      float inv = rsqrtf(var + 1e-5f);
      for (int f = 0; f < Hh; ++f) {
        float ln = ((float)tp[f]-mean)*inv*tlng[f] + tlnb[f];
        tp[f] = (_Float16)(1.f/(1.f+expf(-ln)));
      }
    }
    // ---- hidden[b,t,:] = gate * (allr @ Wproj + b), straight from WMMA acc ----
    {
      const _Float16* Bw = ws + WOFFS[27];
      for (int nt = 0; nt < 16; ++nt) {
        v8f acc;
#pragma unroll
        for (int r = 0; r < 8; ++r) acc[r] = 0.f;
        for (int kc = 0; kc < 8; ++kc) acc = frag_mma(SB, STW, kc, Bw, nt*8 + kc, lane, acc);
#pragma unroll
        for (int r = 0; r < 8; ++r) {
          int row = r + 8*kh;
          int n   = nt*16 + m;
          float g = (float)tpre[row*256 + n];
          out[OUT_HID + ((size_t)(m0+row)*Tt + t)*Hh + n] = g * (acc[r] + tpb[n]);
        }
      }
    }
  }
}

// ---------------------------------------------------------------------------
// Phase 2: heads over hidden (bandwidth-bound elementwise/dot kernels)
// ---------------------------------------------------------------------------
__global__ void head_pred_kernel(PtrTab pt, const float* __restrict__ hid,
                                 float* __restrict__ out)
{
  int i = blockIdx.x*blockDim.x + threadIdx.x;
  if (i >= Bb*(Tt-1)) return;
  int b = i / (Tt-1), t = i % (Tt-1);
  const float* h  = hid + ((size_t)b*Tt + t)*Hh;
  const float* g  = pt.p[I_NEXT_LNG];
  const float* bb = pt.p[I_NEXT_LNB];
  const float* W  = pt.p[I_NEXT_W];
  const float* wb = pt.p[I_NEXT_B];
  float mean = 0.f;
  for (int f = 0; f < Hh; ++f) mean += h[f];
  mean /= (float)Hh;
  float var = 0.f;
  for (int f = 0; f < Hh; ++f) { float d = h[f]-mean; var += d*d; }
  var /= (float)Hh;
  float inv = rsqrtf(var + 1e-5f);
  float acc[INs];
  for (int c = 0; c < INs; ++c) acc[c] = wb[c];
  for (int f = 0; f < Hh; ++f) {
    float ln = (h[f]-mean)*inv*g[f] + bb[f];
    for (int c = 0; c < INs; ++c) acc[c] += ln * W[f*INs + c];
  }
  const float* xn = pt.p[I_X] + ((size_t)b*Tt + (t+1))*INs;
  float s = 0.f;
  for (int c = 0; c < INs; ++c) {
    out[OUT_PRED + ((size_t)b*(Tt-1)+t)*INs + c] = acc[c];
    float d = acc[c]-xn[c]; s += d*d;
  }
  out[OUT_SUR + (size_t)b*(Tt-1) + t] = s / (float)INs;
}

__global__ void head_anom_kernel(PtrTab pt, const float* __restrict__ hid,
                                 float* __restrict__ out)
{
  int i = blockIdx.x*blockDim.x + threadIdx.x;
  if (i >= Bb*Tt) return;
  const float* h  = hid + (size_t)i*Hh;
  const float* g  = pt.p[I_ANOM_LNG];
  const float* bb = pt.p[I_ANOM_LNB];
  const float* W  = pt.p[I_ANOM_W];
  float mean = 0.f;
  for (int f = 0; f < Hh; ++f) mean += h[f];
  mean /= (float)Hh;
  float var = 0.f;
  for (int f = 0; f < Hh; ++f) { float d = h[f]-mean; var += d*d; }
  var /= (float)Hh;
  float inv = rsqrtf(var + 1e-5f);
  float acc = pt.p[I_ANOM_B][0];
  for (int f = 0; f < Hh; ++f)
    acc += ((h[f]-mean)*inv*g[f] + bb[f]) * W[f];
  out[OUT_ANOM + i] = 1.f/(1.f+expf(-acc));
}

__global__ void head_props_kernel(PtrTab pt, const float* __restrict__ hid,
                                  float* __restrict__ out)
{
  int b = threadIdx.x;
  if (b >= Bb) return;
  const float* h  = hid + ((size_t)b*Tt + (Tt-1))*Hh;
  const float* g  = pt.p[I_PROP_LNG];
  const float* bb = pt.p[I_PROP_LNB];
  const float* W1 = pt.p[I_P1_W];
  const float* b1 = pt.p[I_P1_B];
  const float* W2 = pt.p[I_P2_W];
  float mean = 0.f;
  for (int f = 0; f < Hh; ++f) mean += h[f];
  mean /= (float)Hh;
  float var = 0.f;
  for (int f = 0; f < Hh; ++f) { float d = h[f]-mean; var += d*d; }
  var /= (float)Hh;
  float inv = rsqrtf(var + 1e-5f);
  float a0 = pt.p[I_P2_B][0], a1 = pt.p[I_P2_B][1];
  for (int j = 0; j < Hh/2; ++j) {
    float z = b1[j];
    for (int f = 0; f < Hh; ++f)
      z += ((h[f]-mean)*inv*g[f] + bb[f]) * W1[f*(Hh/2) + j];
    z = 0.5f*z*(1.f + erff(z*0.70710678118f));     // exact GELU
    a0 += z*W2[j*2+0];
    a1 += z*W2[j*2+1];
  }
  out[OUT_PROP + b*2 + 0] = a0;
  out[OUT_PROP + b*2 + 1] = a1;
}

// ---------------------------------------------------------------------------
extern "C" void kernel_launch(void* const* d_in, const int* in_sizes, int n_in,
                              void* d_out, int out_size, void* d_ws, size_t ws_size,
                              hipStream_t stream)
{
  (void)in_sizes; (void)out_size; (void)ws_size;
  PtrTab pt;
  for (int i = 0; i < NIN; ++i) pt.p[i] = (i < n_in) ? (const float*)d_in[i] : nullptr;
  _Float16* ws = (_Float16*)d_ws;
  float* out = (float*)d_out;

  pack_weights_kernel<<<NW, 256, 0, stream>>>(pt, ws);

  const size_t smem = (size_t)Bb*STW*2 + 8*16*64*4 + 8*16*128*2 + 8*16*256*2
                    + 64 + 8*16*4*4; // 198720 B
  (void)hipFuncSetAttribute(reinterpret_cast<const void*>(brain_rnn_kernel),
                            hipFuncAttributeMaxDynamicSharedMemorySize, (int)smem);
  brain_rnn_kernel<<<1, 256, smem, stream>>>(pt, ws, out);

  head_pred_kernel <<<(Bb*(Tt-1)+255)/256, 256, 0, stream>>>(pt, out + OUT_HID, out);
  head_anom_kernel <<<(Bb*Tt    +255)/256, 256, 0, stream>>>(pt, out + OUT_HID, out);
  head_props_kernel<<<1, 128, 0, stream>>>(pt, out + OUT_HID, out);
}
